// GraphTransformer_16260746183219
// MI455X (gfx1250) — matrix operand order repaired and checked by reading, not compiled, not run on hardware
//
#include <hip/hip_runtime.h>

#define B 32
#define L 256
#define E 768
#define H 12
#define AH 64
#define NUMF 16
#define E3 (3*E)
#define E4 (4*E)

typedef __attribute__((ext_vector_type(16))) __bf16 v16bf;
typedef __attribute__((ext_vector_type(8)))  float  v8f;
typedef int v4i __attribute__((vector_size(16)));

#ifndef __has_builtin
#define __has_builtin(x) 0
#endif
#if __has_builtin(__builtin_amdgcn_global_load_async_to_lds_b128)
#define HAVE_ASYNC 1
#else
#define HAVE_ASYNC 0
#endif

union FragU { v16bf v; uint4 q[2]; };

__device__ __forceinline__ v8f v8f_zero() {
  v8f z;
  #pragma unroll
  for (int i = 0; i < 8; i++) z[i] = 0.0f;
  return z;
}

__device__ __forceinline__ unsigned short f2bf(float x) {
  unsigned int u = __float_as_uint(x);
  u += 0x7FFFu + ((u >> 16) & 1u);           // round-to-nearest-even
  return (unsigned short)(u >> 16);
}

// A-matrix 16x32 bf16 fragment (ISA 7.12.2): lane holds row M=lane%16,
// e=0..7 -> K = 8*(lane/16)+e ; e=8..15 -> K = 16+8*(lane/16)+(e-8)
__device__ __forceinline__ v16bf frag_ld_a(const unsigned short* p, int hi) {
  FragU f;
  f.q[0] = *reinterpret_cast<const uint4*>(p + 8 * hi);
  f.q[1] = *reinterpret_cast<const uint4*>(p + 16 + 8 * hi);
  return f.v;
}

// B-matrix 32x16 bf16 fragment: lane holds col N=lane%16,
// e=0..7 -> K = 8*(lane/16)+e ; e=8..15 -> K = 16+8*(lane/16)+(e-8)
__device__ __forceinline__ v16bf frag_ld_b(const unsigned short* p, int hi) {
  FragU f;
  f.q[0] = *reinterpret_cast<const uint4*>(p + 8 * hi);
  f.q[1] = *reinterpret_cast<const uint4*>(p + 16 + 8 * hi);
  return f.v;
}

// 16-byte global -> LDS copy; async (ASYNCcnt-tracked DMA) when available.
__device__ __forceinline__ void cp16(unsigned short* dst, const unsigned short* src) {
#if HAVE_ASYNC
  __builtin_amdgcn_global_load_async_to_lds_b128(
      (__attribute__((address_space(1))) v4i*)(void*)src,
      (__attribute__((address_space(3))) v4i*)(void*)dst, 0, 0);
#else
  *reinterpret_cast<uint4*>(dst) = *reinterpret_cast<const uint4*>(src);
#endif
}

__device__ __forceinline__ void wait_async0() {
#if HAVE_ASYNC
#if __has_builtin(__builtin_amdgcn_s_wait_asynccnt)
  __builtin_amdgcn_s_wait_asynccnt(0);
#else
  asm volatile("s_wait_asynccnt 0x0" ::: "memory");
#endif
#endif
}

// fp32 [K][N] -> bf16 transposed [N][K], 32x32 LDS tiles.
__global__ __launch_bounds__(256) void cvtT_k(const float* __restrict__ in,
                                              unsigned short* __restrict__ out,
                                              int K, int N) {
  __shared__ unsigned short tile[32][33];
  int n0 = blockIdx.x * 32, k0 = blockIdx.y * 32;
  int col = threadIdx.x & 31, rg = threadIdx.x >> 5;
  #pragma unroll
  for (int p = 0; p < 4; p++) {
    int k = rg + p * 8;
    tile[k][col] = f2bf(in[(size_t)(k0 + k) * N + n0 + col]);  // coalesced read over n
  }
  __syncthreads();
  #pragma unroll
  for (int p = 0; p < 4; p++) {
    int n = rg + p * 8;
    out[(size_t)(n0 + n) * K + k0 + col] = tile[col][n];       // coalesced write over k
  }
}

__global__ __launch_bounds__(256) void embed_k(
    const float* __restrict__ num, const int* __restrict__ cat,
    const float* __restrict__ Wnum, const float* __restrict__ t0,
    const float* __restrict__ t1, const float* __restrict__ t2,
    float* __restrict__ X) {
  int bl = blockIdx.x;
  int c0 = cat[bl * 3 + 0], c1 = cat[bl * 3 + 1], c2 = cat[bl * 3 + 2];
  const float* nr = num + (size_t)bl * NUMF;
  for (int e = threadIdx.x; e < E; e += 256) {
    float v = t0[c0 * E + e] + t1[c1 * E + e] + t2[c2 * E + e];
    #pragma unroll
    for (int j = 0; j < NUMF; j++) v += nr[j] * Wnum[j * E + e];
    X[(size_t)bl * E + e] = v;
  }
}

// One wave32 per row of E=768.
__global__ __launch_bounds__(256) void layernorm_k(
    const float* __restrict__ X, const float* __restrict__ g, const float* __restrict__ b,
    float* __restrict__ Yf, unsigned short* __restrict__ Yb) {
  int w = threadIdx.x >> 5, lane = threadIdx.x & 31;
  int row = blockIdx.x * 8 + w;
  const float* xr = X + (size_t)row * E;
  float vals[E / 32];
  float s = 0.f, ss = 0.f;
  #pragma unroll
  for (int i = 0; i < E / 32; i++) {
    float v = xr[lane + i * 32];
    vals[i] = v; s += v; ss += v * v;
  }
  #pragma unroll
  for (int o = 16; o > 0; o >>= 1) { s += __shfl_xor(s, o, 32); ss += __shfl_xor(ss, o, 32); }
  float mu = s * (1.0f / E);
  float var = ss * (1.0f / E) - mu * mu;
  float rs = rsqrtf(var + 1e-5f);
  #pragma unroll
  for (int i = 0; i < E / 32; i++) {
    int e = lane + i * 32;
    float v = (vals[i] - mu) * rs * g[e] + b[e];
    Yf[(size_t)row * E + e] = v;
    if (Yb) Yb[(size_t)row * E + e] = f2bf(v);
  }
}

// bf16 GEMM, f32 accumulate: C[M,N] = A[M,K] @ Wt[N,K]^T (+bias,relu,pad,resid)
// block = 256 thr (8 waves), block tile 128x256, wave tile 64x64, K-step 32.
// Double-buffered LDS staging via async global->LDS copies.
__global__ __launch_bounds__(256) void gemm_bf16(
    const unsigned short* __restrict__ Abf, const unsigned short* __restrict__ Wt,
    const float* __restrict__ bias, const float* __restrict__ resid,
    const unsigned char* __restrict__ padrow,
    float* __restrict__ Cf, unsigned short* __restrict__ Cb,
    int M, int N, int K, int relu) {
  __shared__ __align__(16) unsigned short As[2][128 * 32];  // [m][k]
  __shared__ __align__(16) unsigned short Bs[2][256 * 40];  // [n][k], padded stride

  const int t = threadIdx.x;
  const int lane = t & 31, w = t >> 5;
  const int hi = lane >> 4, lo = lane & 15;
  const int wm = w & 1, wn = w >> 1;
  const int m0 = blockIdx.y * 128, n0 = blockIdx.x * 256;
  (void)M;

  v8f acc[4][4];
  #pragma unroll
  for (int i = 0; i < 4; i++)
    #pragma unroll
    for (int j = 0; j < 4; j++) acc[i][j] = v8f_zero();

  auto stage = [&](int buf, int kk) {
    #pragma unroll
    for (int r = 0; r < 2; r++) {                 // A tile: 128x32 = 512 chunks
      int c = t + 256 * r;
      int row = c >> 2, col = (c & 3) * 8;
      cp16(&As[buf][row * 32 + col], Abf + (size_t)(m0 + row) * K + kk + col);
    }
    #pragma unroll
    for (int r = 0; r < 4; r++) {                 // B tile: 256x32 = 1024 chunks
      int c = t + 256 * r;
      int n = c >> 2, col = (c & 3) * 8;
      cp16(&Bs[buf][n * 40 + col], Wt + (size_t)(n0 + n) * K + kk + col);
    }
  };

  stage(0, 0);
  int cur = 0;
  for (int k0 = 0; k0 < K; k0 += 32) {
    wait_async0();
    __syncthreads();                 // buf[cur] ready; all waves done with buf[cur^1]
    if (k0 + 32 < K) stage(cur ^ 1, k0 + 32);

    v16bf af[4], bfr[4];
    #pragma unroll
    for (int mi = 0; mi < 4; mi++)
      af[mi] = frag_ld_a(&As[cur][(wm * 64 + mi * 16 + lo) * 32], hi);
    #pragma unroll
    for (int ni = 0; ni < 4; ni++)
      bfr[ni] = frag_ld_b(&Bs[cur][(wn * 64 + ni * 16 + lo) * 40], hi);
    #pragma unroll
    for (int mi = 0; mi < 4; mi++)
      #pragma unroll
      for (int ni = 0; ni < 4; ni++)
        acc[mi][ni] = __builtin_amdgcn_wmma_f32_16x16x32_bf16(
            false, af[mi], false, bfr[ni], (short)0, acc[mi][ni], false, false);
    cur ^= 1;
  }

  // epilogue: C layout lane: N = lo, M = r + 8*hi
  #pragma unroll
  for (int mi = 0; mi < 4; mi++)
    #pragma unroll
    for (int ni = 0; ni < 4; ni++)
      #pragma unroll
      for (int r = 0; r < 8; r++) {
        int gm = m0 + wm * 64 + mi * 16 + r + 8 * hi;
        int gn = n0 + wn * 64 + ni * 16 + lo;
        float v = acc[mi][ni][r];
        if (bias) v += bias[gn];
        if (relu) v = v > 0.f ? v : 0.f;
        if (padrow && padrow[gm]) v = 0.f;
        if (resid) v += resid[(size_t)gm * N + gn];
        if (Cf) Cf[(size_t)gm * N + gn] = v;
        if (Cb) Cb[(size_t)gm * N + gn] = f2bf(v);
      }
}

// Attention: block = (64 queries) x (one b,h). 8 waves.
#define SSTR 257
#define PSTR 264
__global__ __launch_bounds__(256) void attn_k(
    const unsigned short* __restrict__ qkv, const float* __restrict__ dmat,
    const unsigned char* __restrict__ pad, unsigned short* __restrict__ out, int btype) {
  extern __shared__ char smem[];
  float* S = reinterpret_cast<float*>(smem);                                    // [64][257] f32
  unsigned short* Pb = reinterpret_cast<unsigned short*>(smem + 64 * SSTR * 4); // [64][264] bf16
  unsigned short* Vt = Pb + 64 * PSTR;                                          // [A=64][264] bf16

  const int qt = blockIdx.x, h = blockIdx.y, b = blockIdx.z;
  const int t = threadIdx.x, lane = t & 31, w = t >> 5;
  const int hi = lane >> 4, lo = lane & 15;
  const int wq = w & 3, wn = w >> 2;

  // stage V^T (V[k][a] -> Vt[a][k])
  for (int idx = t; idx < AH * L; idx += 256) {
    int a = idx & (AH - 1), k = idx >> 6;
    Vt[a * PSTR + k] = qkv[(size_t)(b * L + k) * E3 + h * (3 * AH) + 2 * AH + a];
  }

  // phase 1: S = Q @ K^T  (wave: 16 queries x 128 keys)
  v8f sc[8];
  #pragma unroll
  for (int i = 0; i < 8; i++) sc[i] = v8f_zero();
  const int qg0 = qt * 64 + wq * 16;
  #pragma unroll
  for (int kt = 0; kt < 2; kt++) {
    const unsigned short* qp = qkv + (size_t)(b * L + qg0 + lo) * E3 + h * (3 * AH) + kt * 32;
    v16bf qf = frag_ld_a(qp, hi);
    #pragma unroll
    for (int nt = 0; nt < 8; nt++) {
      int key = wn * 128 + nt * 16 + lo;
      const unsigned short* kp = qkv + (size_t)(b * L + key) * E3 + h * (3 * AH) + AH + kt * 32;
      v16bf kf = frag_ld_b(kp, hi);
      sc[nt] = __builtin_amdgcn_wmma_f32_16x16x32_bf16(
          false, qf, false, kf, (short)0, sc[nt], false, false);
    }
  }

  // scale + mask + bias -> S
  const float scale = 0.125f;  // 1/sqrt(64)
  #pragma unroll
  for (int nt = 0; nt < 8; nt++)
    #pragma unroll
    for (int r = 0; r < 8; r++) {
      int ql = wq * 16 + r + 8 * hi;
      int key = wn * 128 + nt * 16 + lo;
      int qg = qt * 64 + ql;
      float dv = dmat[((size_t)b * L + qg) * L + key];
      bool pq = pad[b * L + qg] != 0, pk = pad[b * L + key] != 0;
      bool m; float bi;
      if (btype == 0)      { m = !(dv < 1.5f); bi = 0.f; }                          // 'L'
      else if (btype == 1) { m = pq || pk;     bi = 0.f; }                          // 'G'
      else if (btype == 2) { m = pq || pk || (qg == key); bi = -2.f * __logf(dv); } // 'C'
      else {                                                                        // 'M'
        if (h < H / 2) { m = pq || pk || (qg == key); bi = -2.f * __logf(dv); }
        else           { m = !(dv < 1.5f); bi = 0.f; }
      }
      S[ql * SSTR + key] = m ? -1e30f : (sc[nt][r] * scale + bi);
    }
  __syncthreads();

  // phase 2: softmax, one thread per query row
  if (t < 64) {
    float mx = -3.4e38f;
    for (int k = 0; k < L; k++) mx = fmaxf(mx, S[t * SSTR + k]);
    float sum = 0.f;
    for (int k = 0; k < L; k++) { float e = __expf(S[t * SSTR + k] - mx); S[t * SSTR + k] = e; sum += e; }
    float inv = 1.f / sum;
    for (int k = 0; k < L; k++) Pb[t * PSTR + k] = f2bf(S[t * SSTR + k] * inv);
  }
  __syncthreads();

  // phase 3: O = P @ V  (wave: 16 queries x 32 a-dims)
  const int wn2 = w >> 2;
  v8f oc[2];
  oc[0] = v8f_zero(); oc[1] = v8f_zero();
  #pragma unroll
  for (int kt = 0; kt < 8; kt++) {
    v16bf pf = frag_ld_a(&Pb[(wq * 16 + lo) * PSTR + kt * 32], hi);
    #pragma unroll
    for (int ni = 0; ni < 2; ni++) {
      v16bf vf = frag_ld_b(&Vt[(wn2 * 32 + ni * 16 + lo) * PSTR + kt * 32], hi);
      oc[ni] = __builtin_amdgcn_wmma_f32_16x16x32_bf16(
          false, pf, false, vf, (short)0, oc[ni], false, false);
    }
  }
  #pragma unroll
  for (int ni = 0; ni < 2; ni++)
    #pragma unroll
    for (int r = 0; r < 8; r++) {
      int qg = qt * 64 + wq * 16 + r + 8 * hi;
      int a = wn2 * 32 + ni * 16 + lo;
      out[(size_t)(b * L + qg) * E + h * AH + a] = f2bf(oc[ni][r]);
    }
}

// final: mean over L then @Wout + bout
__global__ __launch_bounds__(256) void head_k(
    const float* __restrict__ Xf, const float* __restrict__ Wout,
    const float* __restrict__ bout, float* __restrict__ out) {
  __shared__ float red[256];
  int b = blockIdx.x, t = threadIdx.x;
  float p = 0.f;
  for (int e = t; e < E; e += 256) {
    float s = 0.f;
    for (int l = 0; l < L; l++) s += Xf[(size_t)(b * L + l) * E + e];
    p += s * (1.0f / L) * Wout[e];
  }
  red[t] = p;
  __syncthreads();
  for (int o = 128; o > 0; o >>= 1) { if (t < o) red[t] += red[t + o]; __syncthreads(); }
  if (t == 0) out[b] = red[0] + bout[0];
}

extern "C" void kernel_launch(void* const* d_in, const int* in_sizes, int n_in,
                              void* d_out, int out_size, void* d_ws, size_t ws_size,
                              hipStream_t stream) {
  (void)in_sizes; (void)n_in; (void)out_size; (void)ws_size;
  const float* num  = (const float*)d_in[0];
  const int*   cat  = (const int*)d_in[1];
  const float* dmat = (const float*)d_in[2];
  const unsigned char* pad = (const unsigned char*)d_in[4];  // bool array
  const float* Wnum = (const float*)d_in[5];
  const float* t0   = (const float*)d_in[6];
  const float* t1   = (const float*)d_in[7];
  const float* t2   = (const float*)d_in[8];
  const float* gf   = (const float*)d_in[59];
  const float* bfp  = (const float*)d_in[60];
  const float* Wout = (const float*)d_in[61];
  const float* bout = (const float*)d_in[62];

  char* ws = (char*)d_ws;
  size_t off = 0;
  auto take = [&](size_t bytes) -> char* {
    off = (off + 255) & ~(size_t)255;
    char* p = ws + off; off += bytes; return p;
  };
  const size_t BL = (size_t)B * L;
  float* xF            = (float*)take(BL * E * 4);           // x / x2 (aliased through the block)
  float* x0F           = (float*)take(BL * E * 4);           // LN output (residual source)
  unsigned short* x0B  = (unsigned short*)take(BL * E * 2);  // LN output bf16
  unsigned short* qkvB = (unsigned short*)take(BL * E3 * 2);
  unsigned short* atB  = (unsigned short*)take(BL * E * 2);
  unsigned short* hB   = (unsigned short*)take(BL * E4 * 2);
  unsigned short *wqkvB[5], *woB[5], *w1B[5], *w2B[5];       // transposed bf16 [N][K]
  for (int i = 0; i < 5; i++) {
    wqkvB[i] = (unsigned short*)take((size_t)E * E3 * 2);
    woB[i]   = (unsigned short*)take((size_t)E * E * 2);
    w1B[i]   = (unsigned short*)take((size_t)E * E4 * 2);
    w2B[i]   = (unsigned short*)take((size_t)E4 * E * 2);
  }

  // weight conversion fp32 [K][N] -> bf16 transposed [N][K] (deterministic, every launch)
  for (int i = 0; i < 5; i++) {
    cvtT_k<<<dim3(E3 / 32, E  / 32), 256, 0, stream>>>((const float*)d_in[9 + i * 10 + 0], wqkvB[i], E,  E3);
    cvtT_k<<<dim3(E  / 32, E  / 32), 256, 0, stream>>>((const float*)d_in[9 + i * 10 + 1], woB[i],   E,  E);
    cvtT_k<<<dim3(E4 / 32, E  / 32), 256, 0, stream>>>((const float*)d_in[9 + i * 10 + 6], w1B[i],   E,  E4);
    cvtT_k<<<dim3(E  / 32, E4 / 32), 256, 0, stream>>>((const float*)d_in[9 + i * 10 + 8], w2B[i],   E4, E);
  }

  embed_k<<<(int)BL, 256, 0, stream>>>(num, cat, Wnum, t0, t1, t2, xF);

  const int btypes[5] = {0, 0, 1, 2, 3};  // "LLGCM"
  const size_t attnSmem = (size_t)64 * SSTR * 4 + 2 * (size_t)64 * PSTR * 2;
  for (int i = 0; i < 5; i++) {
    const float* g1  = (const float*)d_in[9 + i * 10 + 2];
    const float* b1  = (const float*)d_in[9 + i * 10 + 3];
    const float* g2  = (const float*)d_in[9 + i * 10 + 4];
    const float* b2  = (const float*)d_in[9 + i * 10 + 5];
    const float* bm1 = (const float*)d_in[9 + i * 10 + 7];
    const float* bm2 = (const float*)d_in[9 + i * 10 + 9];

    layernorm_k<<<(int)(BL / 8), 256, 0, stream>>>(xF, g1, b1, x0F, x0B);
    gemm_bf16<<<dim3(E3 / 256, BL / 128), 256, 0, stream>>>(
        x0B, wqkvB[i], nullptr, nullptr, nullptr, nullptr, qkvB, (int)BL, E3, E, 0);
    attn_k<<<dim3(L / 64, H, B), 256, attnSmem, stream>>>(qkvB, dmat, pad, atB, btypes[i]);
    gemm_bf16<<<dim3(E / 256, BL / 128), 256, 0, stream>>>(
        atB, woB[i], nullptr, x0F, nullptr, xF, nullptr, (int)BL, E, E, 0);   // x2 = attn@Wo + x0
    layernorm_k<<<(int)(BL / 8), 256, 0, stream>>>(xF, g2, b2, x0F, x0B);     // x2n
    gemm_bf16<<<dim3(E4 / 256, BL / 128), 256, 0, stream>>>(
        x0B, w1B[i], bm1, nullptr, nullptr, nullptr, hB, (int)BL, E4, E, 1);  // relu(x2n@W1+bm1)
    gemm_bf16<<<dim3(E / 256, BL / 128), 256, 0, stream>>>(
        hB, w2B[i], bm2, x0F, pad, xF, nullptr, (int)BL, E, E4, 0);           // x = maskpad(h@W2+bm2)+x2n
  }
  layernorm_k<<<(int)(BL / 8), 256, 0, stream>>>(xF, gf, bfp, x0F, nullptr);
  head_k<<<B, 256, 0, stream>>>(x0F, Wout, bout, (float*)d_out);
}
